// Attention_60644938219874
// MI455X (gfx1250) — matrix operand order repaired and checked by reading, not compiled, not run on hardware
//
#include <hip/hip_runtime.h>

typedef _Float16 v16h __attribute__((ext_vector_type(16)));
typedef _Float16 v8h  __attribute__((ext_vector_type(8)));
typedef _Float16 h4   __attribute__((ext_vector_type(4)));
typedef _Float16 h2   __attribute__((ext_vector_type(2)));
typedef float    v8f  __attribute__((ext_vector_type(8)));
typedef float    v4f  __attribute__((ext_vector_type(4)));

#define S_LEN 4096
#define D_DIM 64
#define KT    64              // keys per tile
#define QT    256             // queries per workgroup (8 waves x 2 q-tiles x 16)
#define NTILE (S_LEN / KT)

__global__ __launch_bounds__(256)
void fa_fwd_f16wmma(const float* __restrict__ Q, const float* __restrict__ K,
                    const float* __restrict__ V, float* __restrict__ O)
{
    // double-buffered K (row-major) and V (transposed) tiles, f16 (32 KB total)
    __shared__ __align__(32) _Float16 Klds[2][KT][D_DIM];
    __shared__ __align__(32) _Float16 VTlds[2][D_DIM][KT];

    const int tid   = threadIdx.x;
    const int wave  = tid >> 5;
    const int lane  = tid & 31;
    const int hlane = lane & 15;
    const bool hi   = lane >= 16;

    const int bh    = blockIdx.x >> 4;   // B*H = 16
    const int qtile = blockIdx.x & 15;   // 16 q-tiles of 256 rows
    const int qb0   = qtile * QT + wave * 16;  // q-tile A rows for this wave
    const int qb1   = qb0 + 128;               // q-tile B rows for this wave

    const float qscale = 0.125f * 1.44269504088896340736f; // 1/sqrt(64) * log2(e)

    const float* Qp = Q + (size_t)bh * S_LEN * D_DIM;
    const float* Kp = K + (size_t)bh * S_LEN * D_DIM;
    const float* Vp = V + (size_t)bh * S_LEN * D_DIM;
    float*       Op = O + (size_t)bh * S_LEN * D_DIM;

    // ---- Q^T as loop-invariant B operands, two query tiles ----
    v16h qbA[2], qbB[2];
    {
        const float* qrA = Qp + (size_t)(qb0 + hlane) * D_DIM + (hi ? 16 : 0);
        const float* qrB = Qp + (size_t)(qb1 + hlane) * D_DIM + (hi ? 16 : 0);
        #pragma unroll
        for (int f = 0; f < 2; ++f)
            #pragma unroll
            for (int i = 0; i < 16; ++i) {
                qbA[f][i] = (_Float16)(qrA[f * 32 + i] * qscale);
                qbB[f][i] = (_Float16)(qrB[f * 32 + i] * qscale);
            }
    }

    v8f oA[4], oB[4];
    #pragma unroll
    for (int g = 0; g < 4; ++g) { oA[g] = (v8f){}; oB[g] = (v8f){}; }
    float mA = -3.0e38f, lA = 0.0f, mB = -3.0e38f, lB = 0.0f;

    // staging mappings: K = 1 key x 16 d per thread; V = 4 keys x 4 d per thread
    const int kkey = tid >> 2;         // 0..63
    const int kdb  = (tid & 3) * 16;   // 0,16,32,48
    const int vkb  = (tid & 15) * 4;   // 0..60 keys
    const int vdb  = (tid >> 4) * 4;   // 0..60 d

    // ---- tile staging helpers ----
    auto load_tile = [&](int kbase, v4f (&nk)[4], v4f (&nv)[4]) {
        const v4f* kg = (const v4f*)(Kp + (size_t)(kbase + kkey) * D_DIM + kdb);
        #pragma unroll
        for (int q = 0; q < 4; ++q) nk[q] = kg[q];
        #pragma unroll
        for (int i = 0; i < 4; ++i)
            nv[i] = *(const v4f*)(Vp + (size_t)(kbase + vkb + i) * D_DIM + vdb);
    };
    auto store_tile = [&](int nb, const v4f (&nk)[4], const v4f (&nv)[4]) {
        #pragma unroll
        for (int q = 0; q < 2; ++q) {
            v8h kh;
            #pragma unroll
            for (int i = 0; i < 4; ++i) {
                kh[i]     = (_Float16)nk[2 * q][i];
                kh[4 + i] = (_Float16)nk[2 * q + 1][i];
            }
            *(v8h*)&Klds[nb][kkey][kdb + 8 * q] = kh;
        }
        #pragma unroll
        for (int j = 0; j < 4; ++j) {   // d = vdb+j, 4 consecutive keys packed
            h4 t;
            t[0] = (_Float16)nv[0][j]; t[1] = (_Float16)nv[1][j];
            t[2] = (_Float16)nv[2][j]; t[3] = (_Float16)nv[3][j];
            *(h4*)&VTlds[nb][vdb + j][vkb] = t;
        }
    };

    // ---- online softmax + P^T B-operand assembly (register-only) ----
    auto softmax = [&](v8f (&c)[4], float& m, float& l, v8f (&o)[4], v16h (&pbv)[2]) {
        float mx = c[0][0];
        #pragma unroll
        for (int t = 0; t < 4; ++t)
            #pragma unroll
            for (int j = (t == 0 ? 1 : 0); j < 8; ++j) mx = fmaxf(mx, c[t][j]);
        mx = fmaxf(mx, __shfl_xor(mx, 16, 32));
        const float mn    = fmaxf(m, mx);
        const float alpha = __builtin_amdgcn_exp2f(m - mn);
        float p[32], rs = 0.0f;
        #pragma unroll
        for (int t = 0; t < 4; ++t)
            #pragma unroll
            for (int j = 0; j < 8; ++j) {
                p[t * 8 + j] = __builtin_amdgcn_exp2f(c[t][j] - mn);
                rs += p[t * 8 + j];
            }
        rs += __shfl_xor(rs, 16, 32);
        m = mn;
        l = l * alpha + rs;
        #pragma unroll
        for (int g = 0; g < 4; ++g) o[g] = o[g] * alpha;
        #pragma unroll
        for (int u = 0; u < 2; ++u) {
            int pbi[8];
            #pragma unroll
            for (int i = 0; i < 4; ++i) {
                h2 t0; t0[0] = (_Float16)p[u*16 + 2*i];     t0[1] = (_Float16)p[u*16 + 2*i + 1];
                h2 t1; t1[0] = (_Float16)p[u*16 + 8 + 2*i]; t1[1] = (_Float16)p[u*16 + 9 + 2*i];
                int a  = __builtin_bit_cast(int, t0);
                int b  = __builtin_bit_cast(int, t1);
                int xa = __shfl_xor(a, 16, 32);
                int xb = __shfl_xor(b, 16, 32);
                pbi[i]     = hi ? xb : a;
                pbi[4 + i] = hi ? b  : xa;
            }
            #pragma unroll
            for (int i = 0; i < 8; ++i) {
                h2 t = __builtin_bit_cast(h2, pbi[i]);
                pbv[u][2 * i] = t[0]; pbv[u][2 * i + 1] = t[1];
            }
        }
    };

    // ---- preload tile 0 ----
    {
        v4f k0[4], v0[4];
        load_tile(0, k0, v0);
        store_tile(0, k0, v0);
    }
    __syncthreads();

    for (int kt = 0; kt < NTILE; ++kt) {
        const int  buf  = kt & 1;
        const bool more = (kt + 1) < NTILE;

        v4f nk[4], nv[4];
        if (more) load_tile((kt + 1) * KT, nk, nv);

        // ---- S^T = K Q^T : K A-frags shared by both query tiles (16 WMMAs) ----
        v8f cA[4], cB[4];
        #pragma unroll
        for (int t = 0; t < 4; ++t) {
            v8f aA = (v8f){}, aB = (v8f){};
            #pragma unroll
            for (int f = 0; f < 2; ++f) {
                const _Float16* kp = &Klds[buf][t * 16 + hlane][f * 32 + (hi ? 16 : 0)];
                v8h r0 = *(const v8h*)kp;
                v8h r1 = *(const v8h*)(kp + 8);
                v16h ka;
                #pragma unroll
                for (int i = 0; i < 8; ++i) { ka[i] = r0[i]; ka[8 + i] = r1[i]; }
                aA = __builtin_amdgcn_wmma_f32_16x16x32_f16(false, ka, false, qbA[f],
                                                            (short)0, aA, false, false);
                aB = __builtin_amdgcn_wmma_f32_16x16x32_f16(false, ka, false, qbB[f],
                                                            (short)0, aB, false, false);
            }
            cA[t] = aA; cB[t] = aB;
        }

        // ---- two independent softmax streams (fills WMMA hazard slots) ----
        v16h pbvA[2], pbvB[2];
        softmax(cA, mA, lA, oA, pbvA);
        softmax(cB, mB, lB, oB, pbvB);

        // ---- O^T += V^T P^T : V^T A-frags shared by both query tiles (16 WMMAs) ----
        #pragma unroll
        for (int g = 0; g < 4; ++g) {
            v8f accA = oA[g], accB = oB[g];
            #pragma unroll
            for (int u = 0; u < 2; ++u) {
                const _Float16* vp = &VTlds[buf][g * 16 + hlane][u * 32 + (hi ? 16 : 0)];
                v8h r0 = *(const v8h*)vp;
                v8h r1 = *(const v8h*)(vp + 8);
                v16h va;
                #pragma unroll
                for (int i = 0; i < 8; ++i) { va[i] = r0[i]; va[8 + i] = r1[i]; }
                accA = __builtin_amdgcn_wmma_f32_16x16x32_f16(false, va, false, pbvA[u],
                                                              (short)0, accA, false, false);
                accB = __builtin_amdgcn_wmma_f32_16x16x32_f16(false, va, false, pbvB[u],
                                                              (short)0, accB, false, false);
            }
            oA[g] = accA; oB[g] = accB;
        }

        if (more) store_tile(buf ^ 1, nk, nv);
        __syncthreads();
    }

    // ---- normalize and store both query tiles ----
    const float invA = 1.0f / lA, invB = 1.0f / lB;
    float* orA = Op + (size_t)(qb0 + hlane) * D_DIM + (hi ? 8 : 0);
    float* orB = Op + (size_t)(qb1 + hlane) * D_DIM + (hi ? 8 : 0);
    #pragma unroll
    for (int g = 0; g < 4; ++g) {
        v8f onA = oA[g] * invA, onB = oB[g] * invB;
        v4f a0, a1, b0, b1;
        #pragma unroll
        for (int i = 0; i < 4; ++i) {
            a0[i] = onA[i]; a1[i] = onA[4 + i];
            b0[i] = onB[i]; b1[i] = onB[4 + i];
        }
        *(v4f*)(orA + g * 16)     = a0;
        *(v4f*)(orA + g * 16 + 4) = a1;
        *(v4f*)(orB + g * 16)     = b0;
        *(v4f*)(orB + g * 16 + 4) = b1;
    }
}

extern "C" void kernel_launch(void* const* d_in, const int* in_sizes, int n_in,
                              void* d_out, int out_size, void* d_ws, size_t ws_size,
                              hipStream_t stream) {
    const float* Q = (const float*)d_in[0];
    const float* K = (const float*)d_in[1];
    const float* V = (const float*)d_in[2];
    float* O = (float*)d_out;
    // grid = B*H * (S / QT) = 16 * 16 = 256 workgroups of 256 threads (8 waves)
    fa_fwd_f16wmma<<<dim3(256), 256, 0, stream>>>(Q, K, V, O);
}
